// multiLayer_TranLSTM_12326556139970
// MI455X (gfx1250) — matrix operand order repaired and checked by reading, not compiled, not run on hardware
//
#include <hip/hip_runtime.h>
#include <hip/hip_bf16.h>

// ---------------------------------------------------------------------------
// MI455X (gfx1250) implementation.
// GEMMs use v_wmma_f32_16x16x32_f16 (wave32). Weights (~35MB f16) and
// activations fit in the 192MB L2, so fragments stream straight from
// global/L2. Wave tile = 32x64 (2 A-frags x 4 B-frags, 8 accumulators):
// each B fragment feeds two WMMAs -> 0.75KB operand traffic per WMMA.
// All in-loop addresses are saddr-form (uniform scalar base advanced by K,
// loop-invariant per-lane i32 offsets) so the inner loop is loads+WMMA only.
// ---------------------------------------------------------------------------

typedef _Float16 h8  __attribute__((ext_vector_type(8)));
typedef _Float16 h16 __attribute__((ext_vector_type(16)));
typedef float    f8  __attribute__((ext_vector_type(8)));

#define DEVINL __device__ __forceinline__

// ---------------- f32 -> f16 conversion ----------------
__global__ void k_cvt_f32_f16(const float* __restrict__ s, _Float16* __restrict__ d, int n) {
  int i = blockIdx.x * blockDim.x + threadIdx.x;
  if (i < n) d[i] = (_Float16)s[i];
}

// ---------------- WMMA GEMM: C[M,N] = act(A[M,K] @ W[N,K]^T + bias) --------
// Wave tile: 32 (M) x 64 (N).  Block: 8 waves stacked in M -> 256 x 64 tile.
// A fragment (16x32 f16, this code's contiguous k order, same as B): lane L holds row M=(L&15);
//   elements 0..7  = K in [k + (L>>4)*16,     +8)
//   elements 8..15 = K in [k + (L>>4)*16 + 8, +8)
// B fragment (32x16 f16): lane L holds column N=(L&15) = row of W;
//   elements 0..15 = K in [k + (L>>4)*16, +16)  -> one contiguous 32B load.
// C/D (16x16 f32): lane L col N=(L&15), VGPR v -> row M = (L>>4)*8 + v.
template<int RELU, int OF32, int OF16>
__global__ void k_gemm(const _Float16* __restrict__ A,
                       const _Float16* __restrict__ W,
                       const float* __restrict__ bias,
                       float* __restrict__ Cf, _Float16* __restrict__ Ch,
                       int M, int N, int K) {
  const int lane = threadIdx.x & 31;
  const int wave = threadIdx.x >> 5;
  const int m0 = (blockIdx.y * 8 + wave) * 32;
  const int n0 = blockIdx.x * 64;
  if (m0 >= M) return;
  const int lr = lane & 15;
  const int hi = lane >> 4;

  f8 acc[2][4];
#pragma unroll
  for (int j = 0; j < 4; ++j) {
    float bv = bias ? bias[n0 + j * 16 + lr] : 0.0f;
#pragma unroll
    for (int v = 0; v < 8; ++v) { acc[0][j][v] = bv; acc[1][j][v] = bv; }
  }

  // Loop-invariant per-lane byte offsets (i32); the K-advance stays in the
  // uniform scalar base pointer -> saddr-form global loads, no in-loop VALU
  // address arithmetic (avoids the F16-WMMA WAR-hazard v_nops).
  const char* aB = (const char*)A;
  const char* wB = (const char*)W;
  const int aOff0 = (int)(((unsigned)(m0 + lr) * (unsigned)K + (unsigned)(hi * 16)) * 2u);
  const int aOff1 = aOff0 + K * 32;           // +16 rows
  const int bOff0 = (int)(((unsigned)(n0 + lr) * (unsigned)K + (unsigned)(hi * 16)) * 2u);
  const int bStride = K * 32;                 // +16 W rows

  for (int k = 0; k < K; k += 32) {
    const char* ak = aB + (size_t)k * 2;
    const char* wk = wB + (size_t)k * 2;
    h8 a0lo = *(const h8*)(ak + aOff0);
    h8 a0hi = *(const h8*)(ak + aOff0 + 16);
    h8 a1lo = *(const h8*)(ak + aOff1);
    h8 a1hi = *(const h8*)(ak + aOff1 + 16);
    h16 bf[4];
#pragma unroll
    for (int j = 0; j < 4; ++j)
      bf[j] = *(const h16*)(wk + (size_t)(bOff0 + j * bStride));
    h16 af0 = __builtin_shufflevector(a0lo, a0hi,
                                      0, 1, 2, 3, 4, 5, 6, 7,
                                      8, 9, 10, 11, 12, 13, 14, 15);
    h16 af1 = __builtin_shufflevector(a1lo, a1hi,
                                      0, 1, 2, 3, 4, 5, 6, 7,
                                      8, 9, 10, 11, 12, 13, 14, 15);
#pragma unroll
    for (int j = 0; j < 4; ++j) {
      acc[0][j] = __builtin_amdgcn_wmma_f32_16x16x32_f16(
          false, af0, false, bf[j], (short)0, acc[0][j], false, false);
      acc[1][j] = __builtin_amdgcn_wmma_f32_16x16x32_f16(
          false, af1, false, bf[j], (short)0, acc[1][j], false, false);
    }
  }

#pragma unroll
  for (int mi = 0; mi < 2; ++mi) {
#pragma unroll
    for (int j = 0; j < 4; ++j) {
#pragma unroll
      for (int v = 0; v < 8; ++v) {
        int r = m0 + mi * 16 + hi * 8 + v;
        int c = n0 + j * 16 + lr;
        float val = acc[mi][j][v];
        if (RELU) val = fmaxf(val, 0.0f);
        if (OF32) Cf[(size_t)r * N + c] = val;
        if (OF16) Ch[(size_t)r * N + c] = (_Float16)val;
      }
    }
  }
}

// ---------------- causal MHA over T=10 (tiny, f32 VALU) ----------------
#define NHEAD_C 16
#define TSEQ 10

__global__ void k_attn(const float* __restrict__ qkv, _Float16* __restrict__ ctx,
                       int B, int H) {
  int tid = blockIdx.x * blockDim.x + threadIdx.x;
  int total = B * NHEAD_C * TSEQ;
  if (tid >= total) return;
  int q  = tid % TSEQ;
  int hh = (tid / TSEQ) % NHEAD_C;
  int b  = tid / (TSEQ * NHEAD_C);
  const int hd = H / NHEAD_C;  // 64
  const float scale = rsqrtf((float)hd);

  size_t rowQ = ((size_t)b * TSEQ + q) * 3 * H + (size_t)hh * hd;
  float s[TSEQ];
  float mx = -1e30f;
  for (int kk = 0; kk <= q; ++kk) {
    size_t rowK = ((size_t)b * TSEQ + kk) * 3 * H + H + (size_t)hh * hd;
    float acc = 0.f;
    for (int d = 0; d < hd; ++d) acc += qkv[rowQ + d] * qkv[rowK + d];
    acc *= scale;
    s[kk] = acc;
    mx = fmaxf(mx, acc);
  }
  float denom = 0.f;
  for (int kk = 0; kk <= q; ++kk) { s[kk] = __expf(s[kk] - mx); denom += s[kk]; }
  float inv = 1.f / denom;
  size_t outRow = ((size_t)b * TSEQ + q) * H + (size_t)hh * hd;
  for (int d = 0; d < hd; ++d) {
    float acc = 0.f;
    for (int kk = 0; kk <= q; ++kk) {
      size_t rowV = ((size_t)b * TSEQ + kk) * 3 * H + 2 * H + (size_t)hh * hd;
      acc += s[kk] * qkv[rowV + d];
    }
    ctx[outRow + d] = (_Float16)(acc * inv);
  }
}

// ---------------- out = LayerNorm(X + A) (one block per row) --------------
__global__ void k_add_ln(const float* __restrict__ X, const float* __restrict__ Aa,
                         const float* __restrict__ w, const float* __restrict__ b,
                         float* __restrict__ of, _Float16* __restrict__ oh, int H) {
  int row = blockIdx.x;
  const float* x = X + (size_t)row * H;
  const float* a = Aa + (size_t)row * H;
  float s = 0.f, s2 = 0.f;
  for (int i = threadIdx.x; i < H; i += blockDim.x) {
    float v = x[i] + a[i];
    s += v; s2 += v * v;
  }
  __shared__ float sb[256], sb2[256];
  sb[threadIdx.x] = s; sb2[threadIdx.x] = s2;
  __syncthreads();
  for (int off = 128; off > 0; off >>= 1) {
    if ((int)threadIdx.x < off) {
      sb[threadIdx.x]  += sb[threadIdx.x + off];
      sb2[threadIdx.x] += sb2[threadIdx.x + off];
    }
    __syncthreads();
  }
  float mean = sb[0] / H;
  float var  = sb2[0] / H - mean * mean;
  float rs = rsqrtf(var + 1e-5f);
  for (int i = threadIdx.x; i < H; i += blockDim.x) {
    float v = (x[i] + a[i] - mean) * rs * w[i] + b[i];
    of[(size_t)row * H + i] = v;
    oh[(size_t)row * H + i] = (_Float16)v;
  }
}

// ---------------- LSTM state init / cell ----------------
__global__ void k_init_state(const float* __restrict__ uv, const float* __restrict__ gv,
                             int layer, int L, int H, int B,
                             _Float16* __restrict__ hS, float* __restrict__ cS) {
  int i = blockIdx.x * blockDim.x + threadIdx.x;
  if (i >= B * H) return;
  int b = i / H, d = i % H;
  size_t src = ((size_t)b * L + layer) * H + d;
  hS[i] = (_Float16)uv[src];
  cS[i] = gv[src];
}

DEVINL float sigf(float x) { return 1.f / (1.f + __expf(-x)); }

__global__ void k_lstm_cell(const float* __restrict__ XG, const float* __restrict__ G,
                            float* __restrict__ C, _Float16* __restrict__ hS,
                            float* __restrict__ loF, _Float16* __restrict__ loH,
                            float* __restrict__ uvOut, float* __restrict__ gvOut,
                            int t, int layer, int B, int H, int L, int T, int last) {
  int i = blockIdx.x * blockDim.x + threadIdx.x;
  if (i >= B * H) return;
  int b = i / H, d = i % H;
  size_t xrow = ((size_t)b * T + t) * 4 * H;
  size_t grow = (size_t)b * 4 * H;
  float gi = sigf(XG[xrow + d]          + G[grow + d]);
  float gf = sigf(XG[xrow + H + d]      + G[grow + H + d]);
  float gg = tanhf(XG[xrow + 2 * H + d] + G[grow + 2 * H + d]);
  float go = sigf(XG[xrow + 3 * H + d]  + G[grow + 3 * H + d]);
  float c = gf * C[i] + gi * gg;
  C[i] = c;
  float h = go * tanhf(c);
  size_t orow = ((size_t)b * T + t) * H + d;
  loF[orow] = h;
  loH[orow] = (_Float16)h;
  hS[i] = (_Float16)h;
  if (last) {
    size_t oidx = ((size_t)b * L + layer) * H + d;
    uvOut[oidx] = h;   // hT
    gvOut[oidx] = c;   // cT
  }
}

// ---------------- final 2-class head + log_softmax ----------------
__global__ void k_head2(const _Float16* __restrict__ O1, const float* __restrict__ W2,
                        const float* __restrict__ B2, float* __restrict__ out,
                        int M, int K) {
  int m = blockIdx.x * blockDim.x + threadIdx.x;
  if (m >= M) return;
  float s0 = B2[0], s1 = B2[1];
  const _Float16* r = O1 + (size_t)m * K;
  for (int k = 0; k < K; ++k) {
    float v = (float)r[k];
    s0 += v * W2[k];
    s1 += v * W2[K + k];
  }
  float mx = fmaxf(s0, s1);
  float l = mx + logf(__expf(s0 - mx) + __expf(s1 - mx));
  out[(size_t)m * 2]     = s0 - l;
  out[(size_t)m * 2 + 1] = s1 - l;
}

// ---------------------------------------------------------------------------
extern "C" void kernel_launch(void* const* d_in, const int* in_sizes, int n_in,
                              void* d_out, int out_size, void* d_ws, size_t ws_size,
                              hipStream_t stream) {
  (void)in_sizes; (void)n_in; (void)out_size; (void)ws_size;
  const int B = 256, T = 10, D = 512, H = 1024, FF = 2048, L = 2;
  const int M = B * T;  // 2560

  const float* x      = (const float*)d_in[0];
  const float* uv_in  = (const float*)d_in[1];
  const float* gv_in  = (const float*)d_in[2];
  const float* fc_w   = (const float*)d_in[3];
  const float* fc_b   = (const float*)d_in[4];
  const float* inp_w  = (const float*)d_in[5];
  const float* inp_b  = (const float*)d_in[6];
  const float* outp_w = (const float*)d_in[7];
  const float* outp_b = (const float*)d_in[8];
  const float* ln1_w  = (const float*)d_in[9];
  const float* ln1_b  = (const float*)d_in[10];
  const float* ff1_w  = (const float*)d_in[11];
  const float* ff1_b  = (const float*)d_in[12];
  const float* ff2_w  = (const float*)d_in[13];
  const float* ff2_b  = (const float*)d_in[14];
  const float* ln2_w  = (const float*)d_in[15];
  const float* ln2_b  = (const float*)d_in[16];
  const float* wih    = (const float*)d_in[17];
  const float* whh    = (const float*)d_in[18];
  const float* bih    = (const float*)d_in[19];
  const float* bhh    = (const float*)d_in[20];
  const float* o1w    = (const float*)d_in[21];
  const float* o1b    = (const float*)d_in[22];
  const float* o2w    = (const float*)d_in[23];
  const float* o2b    = (const float*)d_in[24];

  // bump allocator over d_ws
  char* p = (char*)d_ws;
  auto alloc = [&](size_t bytes) -> void* {
    void* r = (void*)p;
    p += (bytes + 255) & ~(size_t)255;
    return r;
  };

  // f16 buffers
  _Float16* x_h    = (_Float16*)alloc((size_t)M * D * 2);
  _Float16* fcW_h  = (_Float16*)alloc((size_t)H * D * 2);
  _Float16* inpW_h = (_Float16*)alloc((size_t)3 * H * H * 2);
  _Float16* outpW_h= (_Float16*)alloc((size_t)H * H * 2);
  _Float16* ff1W_h = (_Float16*)alloc((size_t)FF * H * 2);
  _Float16* ff2W_h = (_Float16*)alloc((size_t)H * FF * 2);
  _Float16* wihW_h = (_Float16*)alloc((size_t)4 * H * H * 2);
  _Float16* whhW_h = (_Float16*)alloc((size_t)4 * H * H * 2);
  _Float16* o1W_h  = (_Float16*)alloc((size_t)512 * H * 2);
  _Float16* h_h    = (_Float16*)alloc((size_t)M * H * 2);
  _Float16* ctx_h  = (_Float16*)alloc((size_t)M * H * 2);
  _Float16* h1_h   = (_Float16*)alloc((size_t)M * H * 2);
  _Float16* f_h    = (_Float16*)alloc((size_t)M * FF * 2);
  _Float16* enc_h  = (_Float16*)alloc((size_t)M * H * 2);
  _Float16* lo_h   = (_Float16*)alloc((size_t)M * H * 2);
  _Float16* hs_h   = (_Float16*)alloc((size_t)B * H * 2);
  _Float16* o1_h   = (_Float16*)alloc((size_t)M * 512 * 2);
  // f32 buffers
  float* h_f   = (float*)alloc((size_t)M * H * 4);
  float* qkv_f = (float*)alloc((size_t)M * 3 * H * 4);
  float* a_f   = (float*)alloc((size_t)M * H * 4);
  float* h1_f  = (float*)alloc((size_t)M * H * 4);
  float* enc_f = (float*)alloc((size_t)M * H * 4);
  float* xg_f  = (float*)alloc((size_t)M * 4 * H * 4);
  float* g_f   = (float*)alloc((size_t)B * 4 * H * 4);
  float* c_f   = (float*)alloc((size_t)B * H * 4);
  float* lo_f  = (float*)alloc((size_t)M * H * 4);

  auto cvt = [&](const float* s, _Float16* d, size_t n) {
    k_cvt_f32_f16<<<(unsigned)((n + 255) / 256), 256, 0, stream>>>(s, d, (int)n);
  };
  cvt(x, x_h, (size_t)M * D);
  cvt(fc_w, fcW_h, (size_t)H * D);
  cvt(inp_w, inpW_h, (size_t)3 * H * H);
  cvt(outp_w, outpW_h, (size_t)H * H);
  cvt(ff1_w, ff1W_h, (size_t)FF * H);
  cvt(ff2_w, ff2W_h, (size_t)H * FF);
  cvt(wih, wihW_h, (size_t)4 * H * H);
  cvt(whh, whhW_h, (size_t)4 * H * H);
  cvt(o1w, o1W_h, (size_t)512 * H);

  dim3 blk(256);
  const unsigned my = (M + 255) / 256;   // 10  (block covers 256 rows)
  const unsigned by = (B + 255) / 256;   // 1

  // h = relu(x @ fc_w^T + fc_b)
  k_gemm<1, 1, 1><<<dim3(H / 64, my), blk, 0, stream>>>(x_h, fcW_h, fc_b, h_f, h_h, M, H, D);

  float* outp_logp = (float*)d_out;
  float* gv_out = outp_logp + (size_t)M * 2;           // [B,L,H]
  float* uv_out = gv_out + (size_t)B * L * H;          // [B,L,H]

  const float* curF = h_f;
  const _Float16* curH = h_h;

  for (int j = 0; j < L; ++j) {
    // qkv = h @ in_proj^T + b
    k_gemm<0, 1, 0><<<dim3(3 * H / 64, my), blk, 0, stream>>>(curH, inpW_h, inp_b, qkv_f, nullptr, M, 3 * H, H);
    // causal attention -> ctx (f16)
    k_attn<<<(B * NHEAD_C * TSEQ + 255) / 256, blk, 0, stream>>>(qkv_f, ctx_h, B, H);
    // a = ctx @ out_proj^T + b
    k_gemm<0, 1, 0><<<dim3(H / 64, my), blk, 0, stream>>>(ctx_h, outpW_h, outp_b, a_f, nullptr, M, H, H);
    // h1 = LN1(h + a)
    k_add_ln<<<M, blk, 0, stream>>>(curF, a_f, ln1_w, ln1_b, h1_f, h1_h, H);
    // f = relu(h1 @ ff1^T + b)
    k_gemm<1, 0, 1><<<dim3(FF / 64, my), blk, 0, stream>>>(h1_h, ff1W_h, ff1_b, nullptr, f_h, M, FF, H);
    // a = f @ ff2^T + b
    k_gemm<0, 1, 0><<<dim3(H / 64, my), blk, 0, stream>>>(f_h, ff2W_h, ff2_b, a_f, nullptr, M, H, FF);
    // enc = LN2(h1 + a)
    k_add_ln<<<M, blk, 0, stream>>>(h1_f, a_f, ln2_w, ln2_b, enc_f, enc_h, H);
    // xg = enc @ w_ih^T + b_ih  (all timesteps at once)
    k_gemm<0, 1, 0><<<dim3(4 * H / 64, my), blk, 0, stream>>>(enc_h, wihW_h, bih, xg_f, nullptr, M, 4 * H, H);
    // h0/c0 from uv/gv inputs
    k_init_state<<<(B * H + 255) / 256, blk, 0, stream>>>(uv_in, gv_in, j, L, H, B, hs_h, c_f);
    for (int t = 0; t < T; ++t) {
      // g = h @ w_hh^T + b_hh
      k_gemm<0, 1, 0><<<dim3(4 * H / 64, by), blk, 0, stream>>>(hs_h, whhW_h, bhh, g_f, nullptr, B, 4 * H, H);
      k_lstm_cell<<<(B * H + 255) / 256, blk, 0, stream>>>(xg_f, g_f, c_f, hs_h, lo_f, lo_h,
                                                           uv_out, gv_out, t, j, B, H, L, T,
                                                           (t == T - 1) ? 1 : 0);
    }
    curF = lo_f;
    curH = lo_h;
  }

  // o1 = relu(lstm_out @ out1^T + b)
  k_gemm<1, 0, 1><<<dim3(512 / 64, my), blk, 0, stream>>>(lo_h, o1W_h, o1b, nullptr, o1_h, M, 512, H);
  // logits + log_softmax (2 classes)
  k_head2<<<(M + 255) / 256, blk, 0, stream>>>(o1_h, o2w, o2b, outp_logp, M, 512);
}
